// SelfAttention_56126632624914
// MI455X (gfx1250) — matrix operand order repaired and checked by reading, not compiled, not run on hardware
//
#include <hip/hip_runtime.h>
#include <hip/hip_bf16.h>

#define B_ 8
#define T_ 2048
#define D_ 1024
#define H_ 64

typedef __attribute__((ext_vector_type(16))) __bf16 v16bf;
typedef __attribute__((ext_vector_type(8)))  float  v8f;
typedef __attribute__((ext_vector_type(4)))  int    v4i;

#if defined(__has_builtin)
#  if __has_builtin(__builtin_amdgcn_global_load_async_to_lds_b128) && \
      __has_builtin(__builtin_amdgcn_s_wait_asynccnt)
#    define HAVE_ASYNC 1
#  endif
#endif
#ifndef HAVE_ASYNC
#  define HAVE_ASYNC 0
#endif

#if HAVE_ASYNC
typedef __attribute__((address_space(1))) v4i gv4i;  // global v4i
typedef __attribute__((address_space(3))) v4i lv4i;  // LDS v4i
#endif

__device__ __forceinline__ v8f wmma_bf16(v16bf a, v16bf b, v8f c) {
  // D = A(16x32 bf16) x B(32x16 bf16) + C(16x16 f32)
  return __builtin_amdgcn_wmma_f32_16x16x32_bf16(false, a, false, b, (short)0, c,
                                                 false, false);
}

// 16-byte global->LDS transfer (async on CDNA5, sync fallback otherwise)
__device__ __forceinline__ void g2l_b128(const char* src, char* dst) {
#if HAVE_ASYNC
  char* s = const_cast<char*>(src);
  __builtin_amdgcn_global_load_async_to_lds_b128((gv4i*)s, (lv4i*)dst, 0, 0);
#else
  *(int4*)dst = *(const int4*)src;
#endif
}

// ---------------------------------------------------------------------------
// Phase 1: projections. One wave handles 32 rows x 64 cols of one projection.
//   qh, kh stored row-major [B*T, 64] bf16
//   vh stored transposed    [B, 64, T]  bf16  (so PV B-operands load contiguously)
// ---------------------------------------------------------------------------
__global__ __launch_bounds__(128) void proj_kernel(
    const float* __restrict__ q, const float* __restrict__ k,
    const float* __restrict__ v, const float* __restrict__ Wq,
    const float* __restrict__ Wk, const float* __restrict__ Wv,
    __bf16* __restrict__ qh, __bf16* __restrict__ kh,
    __bf16* __restrict__ vhT) {
  const int lane = threadIdx.x & 31;
  const int grp  = lane >> 4;   // lane group 0/1
  const int lr   = lane & 15;
  const int gwave = blockIdx.x * (blockDim.x >> 5) + (threadIdx.x >> 5);
  const int tilesPerMat = (B_ * T_) / 32;  // 512
  const int which = gwave / tilesPerMat;   // 0=q, 1=k, 2=v
  const int tile  = gwave % tilesPerMat;

  const float* x = (which == 0) ? q  : (which == 1) ? k  : v;
  const float* W = (which == 0) ? Wq : (which == 1) ? Wk : Wv;
  const int m0 = tile * 32;

  v8f acc[2][4] = {};

  for (int k0 = 0; k0 < D_; k0 += 32) {
    // B operands: W^T chunk, B[kk][n] = W[n][k0+kk]. Lane holds N=lr, K=h+16*grp.
    v16bf bm[4];
#pragma unroll
    for (int nt = 0; nt < 4; ++nt) {
      const float* wp = W + (size_t)(nt * 16 + lr) * D_ + k0 + grp * 16;
#pragma unroll
      for (int h = 0; h < 16; ++h) bm[nt][h] = (__bf16)wp[h];
    }
#pragma unroll
    for (int mt = 0; mt < 2; ++mt) {
      // A operand: rows m0+mt*16 .. +15, K=k0..k0+31 (contiguous K = 16*grp + h, same as B)
      const float* xp = x + (size_t)(m0 + mt * 16 + lr) * D_ + k0 + grp * 16;
      v16bf am;
#pragma unroll
      for (int h = 0; h < 8; ++h)  am[h] = (__bf16)xp[h];       // K = h + 16*grp
#pragma unroll
      for (int h = 8; h < 16; ++h) am[h] = (__bf16)xp[h];       // K = h + 16*grp
#pragma unroll
      for (int nt = 0; nt < 4; ++nt) acc[mt][nt] = wmma_bf16(am, bm[nt], acc[mt][nt]);
    }
  }

#pragma unroll
  for (int mt = 0; mt < 2; ++mt)
#pragma unroll
    for (int nt = 0; nt < 4; ++nt)
#pragma unroll
      for (int r = 0; r < 8; ++r) {
        const int row = m0 + mt * 16 + r + grp * 8;   // C layout: M = r + 8*grp
        const int col = nt * 16 + lr;                 //           N = lr
        const __bf16 val = (__bf16)acc[mt][nt][r];
        if (which == 2) {
          const int bb = row / T_, tt = row % T_;
          vhT[((size_t)bb * H_ + col) * T_ + tt] = val;
        } else if (which == 0) {
          qh[(size_t)row * H_ + col] = val;
        } else {
          kh[(size_t)row * H_ + col] = val;
        }
      }
}

// ---------------------------------------------------------------------------
// Phase 2: flash attention. Block = 4 waves = 4 consecutive 16-query tiles of
// one batch. All waves share double-buffered K/V chunk tiles staged in LDS via
// async global->LDS loads (ASYNCcnt) -- causal key ranges overlap, so sharing
// cuts L2 traffic 4x.
// ---------------------------------------------------------------------------
__global__ __launch_bounds__(128) void attn_kernel(
    const __bf16* __restrict__ qh, const __bf16* __restrict__ kh,
    const __bf16* __restrict__ vhT, const int* __restrict__ mask,
    float* __restrict__ out) {
  __shared__ __bf16 kbuf[2][32 * 64];  // [key][feat] row-major, 4KB each
  __shared__ __bf16 vbuf[2][64 * 32];  // [feat][key], 4KB each
  __shared__ __bf16 pbuf[4][16 * 32];  // per-wave P staging (C-layout -> A-layout)

  const int tid  = threadIdx.x;
  const int lane = tid & 31;
  const int grp  = lane >> 4;
  const int lr   = lane & 15;
  const int w    = tid >> 5;

  const int b     = blockIdx.x >> 5;          // 8 batches
  const int qt0   = (blockIdx.x & 31) * 4;    // first of 4 query tiles
  const int qbase = (qt0 + w) * 16;           // this wave's query tile
  const int jendW = qbase + 16;               // causal bound for this wave
  const int nchunk = (qt0 * 16 + 64) / 32;    // block covers keys [0, qt0*16+64)

  const __bf16* khb = kh  + (size_t)b * T_ * H_;
  const __bf16* vtb = vhT + (size_t)b * H_ * T_;
  const int*    mb  = mask + b * T_;

  // Q tile (16x64) as two 16x32 A operands, K split [0,32) and [32,64)
  v16bf aq[2];
  {
    const __bf16* qp = qh + ((size_t)b * T_ + qbase + lr) * H_;
#pragma unroll
    for (int kk = 0; kk < 2; ++kk) {
      const __bf16* p = qp + kk * 32 + grp * 16;
#pragma unroll
      for (int h = 0; h < 8; ++h)  aq[kk][h] = p[h];
#pragma unroll
      for (int h = 8; h < 16; ++h) aq[kk][h] = p[h];
    }
  }

  v8f o[4] = {};
  float mrun[8], lrun[8];
#pragma unroll
  for (int r = 0; r < 8; ++r) { mrun[r] = -1.0e30f; lrun[r] = 0.0f; }

  // ---- stage key chunk j0 (32 keys) into LDS buffer `buf` (128 threads coop)
  auto stage = [&](int j0, int buf) {
    // K tile: kh rows j0..j0+31 are one contiguous 4KB block
    {
      const char* src = (const char*)(khb + (size_t)j0 * H_) + tid * 32;
      char* dst = (char*)&kbuf[buf][0] + tid * 32;
      g2l_b128(src, dst);
      g2l_b128(src + 16, dst + 16);
    }
    // V tile: 64 feature-rows of 64B (stride T_*2 in vhT)
    {
      const int col = tid >> 1, half = tid & 1;
      const char* src =
          (const char*)(vtb + (size_t)col * T_ + j0) + half * 32;
      char* dst = (char*)&vbuf[buf][col * 32] + half * 32;
      g2l_b128(src, dst);
      g2l_b128(src + 16, dst + 16);
    }
  };

  stage(0, 0);

  for (int c = 0; c < nchunk; ++c) {
    const int j0  = c * 32;
    const int cur = c & 1;
#if HAVE_ASYNC
    __builtin_amdgcn_s_wait_asynccnt(0);
#endif
    __syncthreads();  // buffer `cur` visible to all waves; prior reads of `!cur` done
    if (c + 1 < nchunk) stage(j0 + 32, cur ^ 1);

    if (j0 < jendW) {  // wave-uniform causal skip: EXEC stays all-ones inside
      const __bf16* kb = &kbuf[cur][0];
      const __bf16* vb = &vbuf[cur][0];

      // ---- S = Q K^T for two 16-key tiles (B operands from LDS)
      v8f s[2];
#pragma unroll
      for (int t = 0; t < 2; ++t) {
        const __bf16* kp = kb + (t * 16 + lr) * H_ + grp * 16;
        const v16bf b0 = *(const v16bf*)(kp);       // features [0,32)
        const v16bf b1 = *(const v16bf*)(kp + 32);  // features [32,64)
        v8f cc = {};
        cc = wmma_bf16(aq[0], b0, cc);
        cc = wmma_bf16(aq[1], b1, cc);
        s[t] = cc;
      }
      const int km0 = mb[j0 + lr];
      const int km1 = mb[j0 + 16 + lr];

      // ---- scale, mask, online softmax (row stats across 16-lane halves)
      float p0[8], p1[8];
#pragma unroll
      for (int r = 0; r < 8; ++r) {
        const int qrow = qbase + r + grp * 8;
        float s0 = s[0][r] * 0.125f;  // HEAD_SIZE^-0.5
        float s1 = s[1][r] * 0.125f;
        if ((j0 + lr)      > qrow || km0 == 0) s0 = -1.0e30f;
        if ((j0 + 16 + lr) > qrow || km1 == 0) s1 = -1.0e30f;
        float rm = fmaxf(s0, s1);
        rm = fmaxf(rm, __shfl_xor(rm, 1, 32));
        rm = fmaxf(rm, __shfl_xor(rm, 2, 32));
        rm = fmaxf(rm, __shfl_xor(rm, 4, 32));
        rm = fmaxf(rm, __shfl_xor(rm, 8, 32));
        const float mnew  = fmaxf(mrun[r], rm);
        const float alpha = __expf(mrun[r] - mnew);
        mrun[r] = mnew;
        const float e0 = __expf(s0 - mnew);
        const float e1 = __expf(s1 - mnew);
        float rs = e0 + e1;
        rs += __shfl_xor(rs, 1, 32);
        rs += __shfl_xor(rs, 2, 32);
        rs += __shfl_xor(rs, 4, 32);
        rs += __shfl_xor(rs, 8, 32);
        lrun[r] = lrun[r] * alpha + rs;
        p0[r] = e0;
        p1[r] = e1;
#pragma unroll
        for (int nt = 0; nt < 4; ++nt) o[nt][r] *= alpha;
      }

      // ---- P (16x32, C layout f32) -> LDS row-major bf16 -> A operand layout
      __bf16* pb = pbuf[w];
#pragma unroll
      for (int r = 0; r < 8; ++r) {
        const int m = r + grp * 8;
        pb[m * 32 + lr]      = (__bf16)p0[r];
        pb[m * 32 + 16 + lr] = (__bf16)p1[r];
      }
      v16bf pa;
#pragma unroll
      for (int h = 0; h < 8; ++h)  pa[h] = pb[lr * 32 + grp * 16 + h];
#pragma unroll
      for (int h = 8; h < 16; ++h) pa[h] = pb[lr * 32 + grp * 16 + h];

      // ---- O += P V  (B operands from LDS, transposed V layout)
#pragma unroll
      for (int nt = 0; nt < 4; ++nt) {
        const __bf16* vp = vb + (nt * 16 + lr) * 32 + grp * 16;
        const v16bf bv = *(const v16bf*)vp;
        o[nt] = wmma_bf16(pa, bv, o[nt]);
      }
    }
  }

  // ---- normalize and store fp32 output [B,T,64]
#pragma unroll
  for (int r = 0; r < 8; ++r) {
    const float inv = 1.0f / lrun[r];
    const size_t row = (size_t)b * T_ + qbase + r + grp * 8;
#pragma unroll
    for (int nt = 0; nt < 4; ++nt)
      out[row * H_ + nt * 16 + lr] = o[nt][r] * inv;
  }
}

// ---------------------------------------------------------------------------
extern "C" void kernel_launch(void* const* d_in, const int* in_sizes, int n_in,
                              void* d_out, int out_size, void* d_ws, size_t ws_size,
                              hipStream_t stream) {
  (void)in_sizes; (void)n_in; (void)out_size; (void)ws_size;
  const float* k    = (const float*)d_in[0];
  const float* q    = (const float*)d_in[1];
  const float* v    = (const float*)d_in[2];
  const int*   mask = (const int*)d_in[3];
  const float* Wk   = (const float*)d_in[4];
  const float* Wq   = (const float*)d_in[5];
  const float* Wv   = (const float*)d_in[6];
  float* out = (float*)d_out;

  const size_t projBytes = (size_t)B_ * T_ * H_ * sizeof(__bf16);  // 2 MB each
  char* ws = (char*)d_ws;
  __bf16* qh  = (__bf16*)(ws);
  __bf16* kh  = (__bf16*)(ws + projBytes);
  __bf16* vhT = (__bf16*)(ws + 2 * projBytes);

  // Phase 1: 3 projections * 512 row-tiles = 1536 waves, 4 waves/block
  proj_kernel<<<384, 128, 0, stream>>>(q, k, v, Wq, Wk, Wv, qh, kh, vhT);
  // Phase 2: 8 batches * 32 blocks (4 query tiles each), 128 threads
  attn_kernel<<<256, 128, 0, stream>>>(qh, kh, vhT, mask, out);
}
